// CrossAttentionLayer_30064771072613
// MI455X (gfx1250) — matrix operand order repaired and checked by reading, not compiled, not run on hardware
//
#include <hip/hip_runtime.h>
#include <math.h>

typedef __attribute__((ext_vector_type(16))) __bf16 v16bf;
typedef __attribute__((ext_vector_type(8)))  __bf16 v8bf;
typedef __attribute__((ext_vector_type(8)))  float  v8f;

#define NSP 4096   // H*W
#define CIN 256    // input channels
#define BATCH 4
#define DQK 32

__device__ __forceinline__ v8f wmma_bf16(v16bf a, v16bf b, v8f c) {
  // D = A(16x32 bf16) * B(32x16 bf16) + C(16x16 f32)
  return __builtin_amdgcn_wmma_f32_16x16x32_bf16(false, a, false, b, (short)0, c,
                                                 false, false);
}

__device__ __forceinline__ v8f v8f_zero() {
  v8f z;
#pragma unroll
  for (int i = 0; i < 8; ++i) z[i] = 0.0f;
  return z;
}

// ---------------- f32 -> bf16 elementwise convert ----------------
__global__ void cvt_kernel(const float* __restrict__ s, __bf16* __restrict__ d, int n) {
  int i = blockIdx.x * blockDim.x + threadIdx.x;
  if (i < n) d[i] = (__bf16)s[i];
}

// ------------- transpose + convert: x[b][c][n] f32 -> xt[b][n][c] bf16 -------------
__global__ __launch_bounds__(256) void xpose_kernel(const float* __restrict__ X,
                                                    __bf16* __restrict__ Xt) {
  __shared__ float tile[32][33];
  int tx = threadIdx.x, ty = threadIdx.y;         // blockDim = (32, 8)
  int n0 = blockIdx.x * 32, c0 = blockIdx.y * 32, b = blockIdx.z;
  const float* Xb = X + (size_t)b * CIN * NSP;
#pragma unroll
  for (int j = 0; j < 32; j += 8)
    tile[ty + j][tx] = Xb[(size_t)(c0 + ty + j) * NSP + n0 + tx];
  __syncthreads();
  __bf16* Yb = Xt + (size_t)b * NSP * CIN;
#pragma unroll
  for (int j = 0; j < 32; j += 8)
    Yb[(size_t)(n0 + ty + j) * CIN + c0 + tx] = (__bf16)tile[tx][ty + j];
}

// ------------- WMMA projection: Yt[n][c] = BN( sum_k Xt[n][k] * W[c][k] ) -------------
// store_nm=1 -> Y[b][n][Mout] (Q/K layout), store_nm=0 -> Y[b][c][NSP] (V^T layout)
__global__ __launch_bounds__(256) void proj_kernel(
    const __bf16* __restrict__ Xt, const __bf16* __restrict__ Wb,
    const float* __restrict__ cbias, const float* __restrict__ gamma,
    const float* __restrict__ beta, const float* __restrict__ mean,
    const float* __restrict__ var, float extra_scale, int Mout, int store_nm,
    __bf16* __restrict__ Y) {
  const int lane = threadIdx.x & 31;
  const int w    = threadIdx.x >> 5;
  const int wid  = blockIdx.x * 8 + w;
  const int ntiles = NSP / 16;                    // 256
  const int nt = wid % ntiles;
  const int rest = wid / ntiles;
  const int ct = rest % (Mout / 16);
  const int b  = rest / (Mout / 16);
  const int n0 = nt * 16, c0 = ct * 16;
  const int r  = lane & 15, hi = lane >> 4;

  const __bf16* Xb = Xt + (size_t)b * NSP * CIN;
  const __bf16* ap = Xb + (size_t)(n0 + r) * CIN;       // A row (n)
  const __bf16* bp = Wb + (size_t)(c0 + r) * CIN + hi * 16;  // B column (c)
  const int d0 = hi ? 16 : 0, d1 = hi ? 24 : 8;

  v8f acc = v8f_zero();
#pragma unroll
  for (int kc = 0; kc < CIN; kc += 32) {
    v16bf a;
    v8bf lo = *(const v8bf*)(ap + kc + d0);
    v8bf h2 = *(const v8bf*)(ap + kc + d1);
#pragma unroll
    for (int i = 0; i < 8; ++i) { a[i] = lo[i]; a[i + 8] = h2[i]; }
    v16bf bf = *(const v16bf*)(bp + kc);
    acc = wmma_bf16(a, bf, acc);
  }

  const int c = c0 + r;
  float inv = gamma[c] * rsqrtf(var[c] + 1e-5f);
  float sc  = inv * extra_scale;
  float bi  = (cbias[c] * inv + beta[c] - mean[c] * inv) * extra_scale;
#pragma unroll
  for (int i = 0; i < 8; ++i) {
    int n = n0 + i + 8 * hi;
    float y = acc[i] * sc + bi;
    if (store_nm) Y[((size_t)b * NSP + n) * Mout + c] = (__bf16)y;
    else          Y[((size_t)b * Mout + c) * NSP + n] = (__bf16)y;
  }
}

// ------------- fused flash attention: softmax(Q K^T) V^T, out f32 [B][C][N] -------------
__global__ __launch_bounds__(256) void attn_kernel(const __bf16* __restrict__ Q,
                                                   const __bf16* __restrict__ K,
                                                   const __bf16* __restrict__ V,
                                                   float* __restrict__ out) {
  __shared__ float  sS[8][16][32];
  __shared__ __align__(64) __bf16 sP[8][16][32];
  __shared__ float  sA[8][16];
  __shared__ float  sR[8][16];

  const int lane = threadIdx.x & 31;
  const int w    = threadIdx.x >> 5;
  const int b    = blockIdx.x >> 5;               // 32 blocks per batch
  const int mblk = blockIdx.x & 31;
  const int m0   = mblk * 128 + w * 16;
  const int r    = lane & 15, hi = lane >> 4;
  const int d0   = hi ? 16 : 0, d1 = hi ? 24 : 8;

  const __bf16* Qb = Q + (size_t)b * NSP * DQK;
  const __bf16* Kb = K + (size_t)b * NSP * DQK;
  const __bf16* Vb = V + (size_t)b * CIN * NSP;

  // Q A-fragment for this wave's 16 rows (scale already folded in)
  v16bf qf;
  {
    const __bf16* p = Qb + (size_t)(m0 + r) * DQK;
    v8bf lo = *(const v8bf*)(p + d0);
    v8bf h2 = *(const v8bf*)(p + d1);
#pragma unroll
    for (int i = 0; i < 8; ++i) { qf[i] = lo[i]; qf[i + 8] = h2[i]; }
  }

  v8f acc[16];
#pragma unroll
  for (int t = 0; t < 16; ++t) acc[t] = v8f_zero();
  float rmax = -3.0e38f, rsum = 0.0f;

  for (int nb = 0; nb < NSP; nb += 32) {
    // ---- scores: S[16 x 32] = Q * K^T for this key chunk ----
    v16bf k0 = *(const v16bf*)(Kb + (size_t)(nb + r) * DQK + hi * 16);
    v16bf k1 = *(const v16bf*)(Kb + (size_t)(nb + 16 + r) * DQK + hi * 16);
    v8f s0 = wmma_bf16(qf, k0, v8f_zero());
    v8f s1 = wmma_bf16(qf, k1, v8f_zero());
#pragma unroll
    for (int i = 0; i < 8; ++i) {
      sS[w][i + 8 * hi][r]      = s0[i];
      sS[w][i + 8 * hi][r + 16] = s1[i];
    }
    __syncthreads();

    // ---- online softmax: lane owns row r, half hi (16 cols) ----
    float sv[16];
    float mloc = -3.0e38f;
#pragma unroll
    for (int j = 0; j < 16; ++j) {
      sv[j] = sS[w][r][hi * 16 + j];
      mloc = fmaxf(mloc, sv[j]);
    }
    float moth = __shfl_xor(mloc, 16, 32);
    float mnew = fmaxf(rmax, fmaxf(mloc, moth));
    float alpha = __expf(rmax - mnew);
    float psum = 0.0f;
#pragma unroll
    for (int j = 0; j < 16; ++j) {
      float p = __expf(sv[j] - mnew);
      psum += p;
      sP[w][r][hi * 16 + j] = (__bf16)p;
    }
    psum += __shfl_xor(psum, 16, 32);
    rsum = rsum * alpha + psum;
    rmax = mnew;
    if (hi == 0) sA[w][r] = alpha;
    __syncthreads();

    // ---- P A-fragment + per-row rescale factors ----
    v16bf pf;
    {
      const __bf16* p = &sP[w][r][0];
      v8bf lo = *(const v8bf*)(p + d0);
      v8bf h2 = *(const v8bf*)(p + d1);
#pragma unroll
      for (int i = 0; i < 8; ++i) { pf[i] = lo[i]; pf[i + 8] = h2[i]; }
    }
    float al[8];
#pragma unroll
    for (int i = 0; i < 8; ++i) al[i] = sA[w][i + 8 * hi];

    // ---- O += P * V^T over all 256 output channels (16 tiles) ----
#pragma unroll
    for (int t = 0; t < 16; ++t) {
      v8f c;
#pragma unroll
      for (int i = 0; i < 8; ++i) c[i] = acc[t][i] * al[i];
      v16bf vf = *(const v16bf*)(Vb + (size_t)(t * 16 + r) * NSP + nb + hi * 16);
      acc[t] = wmma_bf16(pf, vf, c);
    }
  }

  // ---- normalize and store: out[b][c][m] f32 ----
  if (hi == 0) sR[w][r] = rsum;
  __syncthreads();
  float rs[8];
#pragma unroll
  for (int i = 0; i < 8; ++i) rs[i] = 1.0f / sR[w][i + 8 * hi];
#pragma unroll
  for (int t = 0; t < 16; ++t) {
#pragma unroll
    for (int i = 0; i < 8; ++i) {
      int c = t * 16 + r;
      int m = m0 + i + 8 * hi;
      out[((size_t)b * CIN + c) * NSP + m] = acc[t][i] * rs[i];
    }
  }
}

extern "C" void kernel_launch(void* const* d_in, const int* in_sizes, int n_in,
                              void* d_out, int out_size, void* d_ws, size_t ws_size,
                              hipStream_t stream) {
  const float* x1 = (const float*)d_in[0];
  const float* x2 = (const float*)d_in[1];
  const float* q_w = (const float*)d_in[2];
  const float* q_b = (const float*)d_in[3];
  const float* q_gamma = (const float*)d_in[4];
  const float* q_beta  = (const float*)d_in[5];
  const float* q_mean  = (const float*)d_in[6];
  const float* q_var   = (const float*)d_in[7];
  const float* k_w = (const float*)d_in[8];
  const float* k_b = (const float*)d_in[9];
  const float* k_gamma = (const float*)d_in[10];
  const float* k_beta  = (const float*)d_in[11];
  const float* k_mean  = (const float*)d_in[12];
  const float* k_var   = (const float*)d_in[13];
  const float* v_w = (const float*)d_in[14];
  const float* v_b = (const float*)d_in[15];
  const float* v_gamma = (const float*)d_in[16];
  const float* v_beta  = (const float*)d_in[17];
  const float* v_mean  = (const float*)d_in[18];
  const float* v_var   = (const float*)d_in[19];
  float* out = (float*)d_out;

  // workspace carve-up (all offsets 256B aligned)
  uint8_t* ws = (uint8_t*)d_ws;
  __bf16* x1t = (__bf16*)ws;  ws += (size_t)BATCH * NSP * CIN * 2;   // 8 MB
  __bf16* x2t = (__bf16*)ws;  ws += (size_t)BATCH * NSP * CIN * 2;   // 8 MB
  __bf16* qwb = (__bf16*)ws;  ws += (size_t)DQK * CIN * 2;           // 16 KB
  __bf16* kwb = (__bf16*)ws;  ws += (size_t)DQK * CIN * 2;           // 16 KB
  __bf16* vwb = (__bf16*)ws;  ws += (size_t)CIN * CIN * 2;           // 128 KB
  __bf16* Qb  = (__bf16*)ws;  ws += (size_t)BATCH * NSP * DQK * 2;   // 1 MB
  __bf16* Kb  = (__bf16*)ws;  ws += (size_t)BATCH * NSP * DQK * 2;   // 1 MB
  __bf16* Vt  = (__bf16*)ws;  ws += (size_t)BATCH * CIN * NSP * 2;   // 8 MB

  // 1) weight converts
  cvt_kernel<<<(DQK * CIN + 255) / 256, 256, 0, stream>>>(q_w, qwb, DQK * CIN);
  cvt_kernel<<<(DQK * CIN + 255) / 256, 256, 0, stream>>>(k_w, kwb, DQK * CIN);
  cvt_kernel<<<(CIN * CIN + 255) / 256, 256, 0, stream>>>(v_w, vwb, CIN * CIN);

  // 2) activation transpose+convert: [B][C][N] f32 -> [B][N][C] bf16
  dim3 xg(NSP / 32, CIN / 32, BATCH), xb(32, 8);
  xpose_kernel<<<xg, xb, 0, stream>>>(x1, x1t);
  xpose_kernel<<<xg, xb, 0, stream>>>(x2, x2t);

  // 3) WMMA projections with BN folded (attention scale folded into Q)
  const float qscale = 0.17677669529663687f;  // 1/sqrt(32)
  int qk_blocks = (BATCH * (DQK / 16) * (NSP / 16)) / 8;    // 256
  int v_blocks  = (BATCH * (CIN / 16) * (NSP / 16)) / 8;    // 2048
  proj_kernel<<<qk_blocks, 256, 0, stream>>>(x1t, qwb, q_b, q_gamma, q_beta, q_mean,
                                             q_var, qscale, DQK, 1, Qb);
  proj_kernel<<<qk_blocks, 256, 0, stream>>>(x2t, kwb, k_b, k_gamma, k_beta, k_mean,
                                             k_var, 1.0f, DQK, 1, Kb);
  proj_kernel<<<v_blocks, 256, 0, stream>>>(x2t, vwb, v_b, v_gamma, v_beta, v_mean,
                                            v_var, 1.0f, CIN, 0, Vt);

  // 4) fused flash attention
  attn_kernel<<<BATCH * (NSP / 128), 256, 0, stream>>>(Qb, Kb, Vt, out);
}
